// LieDenseNet_27642409517649
// MI455X (gfx1250) — matrix operand order repaired and checked by reading, not compiled, not run on hardware
//
#include <hip/hip_runtime.h>
#include <math.h>

// ---------------------------------------------------------------------------
// LieConv DenseNet for MI455X (gfx1250, wave32).
// Both contraction stages (PointConv einsum and the Wl linear) are mapped onto
// v_wmma_f32_16x16x32_f16 (16x16 f32 accum, K=32 f16).  MID=16 and K(nn)=32
// match the WMMA tile exactly; every channel count in the network is a
// multiple of 16, so no edge tiles exist.  The Wl GEMM stages its shared
// B k-panels into LDS with gfx1250 async global->LDS copies (ASYNCcnt),
// double-buffered so the next panel is in flight under the current WMMAs.
// ---------------------------------------------------------------------------

typedef __attribute__((ext_vector_type(16))) _Float16 v16h;
typedef __attribute__((ext_vector_type(8)))  _Float16 v8h;
typedef __attribute__((ext_vector_type(8)))  float    v8f;

#define BATCH   4
#define NPER    2048
#define NPTS    (BATCH * NPER)     // 8192
#define KNN     32
#define MIDW    16
#define HIDW    32
#define NFMAX   224                // final channel count; also feat row stride
#define CHUNK   2048               // points per pen/gemm pass (4 chunks)
#define BIGF    3.0e38f

__device__ __forceinline__ float swishf(float x) {
    return x * (1.0f / (1.0f + __expf(-x)));
}

// Async global->LDS copy, 8 bytes per lane (GLOBAL_LOAD_ASYNC_TO_LDS_B64,
// tracked by ASYNCcnt).  lds_off is the 32-bit LDS byte offset (low 32 bits
// of the generic pointer to a __shared__ object).
__device__ __forceinline__ void async_copy_b64(unsigned lds_off,
                                               const float* gsrc) {
    asm volatile("global_load_async_to_lds_b64 %0, %1, off"
                 :: "v"(lds_off), "v"(gsrc)
                 : "memory");
}

// ---------------------------------------------------------------------------
// KNN: per query point, select 32 nearest (masked) neighbors of its batch.
// Outputs: idx (global row), rel (neighbor - self coords), nmask, cnt.
// Neighbor order is irrelevant downstream (all reductions sum over k).
// ---------------------------------------------------------------------------
__global__ __launch_bounds__(128)
void liedn_knn_kernel(const float* __restrict__ coords,
                      const unsigned char* __restrict__ mask,
                      float* __restrict__ rel, int* __restrict__ idx,
                      float* __restrict__ nmask, float* __restrict__ cnt)
{
    __shared__ float sx[128], sy[128], sz[128], sm[128];
    __shared__ float topd[128][KNN];
    __shared__ int   topi[128][KNN];

    const int t = threadIdx.x;
    const int g = blockIdx.x * 128 + t;          // global query row
    const int b = g / NPER;

    const float qx = coords[g * 3 + 0];
    const float qy = coords[g * 3 + 1];
    const float qz = coords[g * 3 + 2];
    const float qm = mask[g] ? 1.0f : 0.0f;

    #pragma unroll
    for (int k = 0; k < KNN; ++k) { topd[t][k] = BIGF; topi[t][k] = b * NPER; }
    float worst = BIGF; int wpos = 0;

    for (int tile = 0; tile < NPER / 128; ++tile) {
        const int j = b * NPER + tile * 128 + t;
        __syncthreads();
        sx[t] = coords[j * 3 + 0];
        sy[t] = coords[j * 3 + 1];
        sz[t] = coords[j * 3 + 2];
        sm[t] = mask[j] ? 1.0f : 0.0f;
        __syncthreads();
        for (int u = 0; u < 128; ++u) {
            const float dx = sx[u] - qx, dy = sy[u] - qy, dz = sz[u] - qz;
            float d2 = dx * dx + dy * dy + dz * dz;
            d2 = (sm[u] != 0.0f) ? d2 : BIGF;
            if (d2 < worst) {
                topd[t][wpos] = d2;
                topi[t][wpos] = b * NPER + tile * 128 + u;
                worst = -1.0f;
                #pragma unroll
                for (int kk = 0; kk < KNN; ++kk) {
                    const float v = topd[t][kk];
                    if (v > worst) { worst = v; wpos = kk; }
                }
            }
        }
    }

    float csum = 0.0f;
    #pragma unroll
    for (int k = 0; k < KNN; ++k) {
        const int jg = topi[t][k];
        const float jm = mask[jg] ? 1.0f : 0.0f;
        const float nmv = jm * qm;
        idx[g * KNN + k]   = jg;
        nmask[g * KNN + k] = nmv;
        rel[(g * KNN + k) * 3 + 0] = coords[jg * 3 + 0] - qx;
        rel[(g * KNN + k) * 3 + 1] = coords[jg * 3 + 1] - qy;
        rel[(g * KNN + k) * 3 + 2] = coords[jg * 3 + 2] - qz;
        csum += nmv;
    }
    cnt[g] = fmaxf(csum, 1.0f);
}

// ---------------------------------------------------------------------------
// PointConv stage: one wave per point.
//   lane l computes kern row for neighbor l (WeightNet swish MLP, abq rows
//   0..2 are zero so only W1 rows 3..5 contribute), stages it in LDS as f16.
//   The wave then forms A = kern^T (16x32) per the ISA f16 A-layout and, per
//   16-channel tile, gathers masked neighbor features as B (32x16, lane = N
//   col) and runs v_wmma_f32_16x16x32_f16.  pen written f16 as [MID*cin].
// ---------------------------------------------------------------------------
__global__ __launch_bounds__(256)
void liedn_pen_kernel(const float* __restrict__ src, int src_stride, int cin,
                      int relu_on,
                      const float* __restrict__ rel,
                      const int* __restrict__ idx,
                      const float* __restrict__ nmask,
                      const float* __restrict__ W1, const float* __restrict__ b1,
                      const float* __restrict__ W2, const float* __restrict__ b2,
                      _Float16* __restrict__ xpen, int chunk_base)
{
    __shared__ _Float16 lk[8][KNN * MIDW];          // per-wave kern tile (f16)

    const int wave = threadIdx.x >> 5;
    const int lane = threadIdx.x & 31;
    const int p = chunk_base + blockIdx.x * 8 + wave;   // global point
    const int l = lane;                                  // neighbor row

    // --- WeightNet kernel MLP for neighbor l ------------------------------
    const float rx = rel[(p * KNN + l) * 3 + 0];
    const float ry = rel[(p * KNN + l) * 3 + 1];
    const float rz = rel[(p * KNN + l) * 3 + 2];

    float kacc[MIDW];
    #pragma unroll
    for (int m = 0; m < MIDW; ++m) kacc[m] = 0.0f;

    #pragma unroll 4
    for (int j = 0; j < HIDW; ++j) {
        float hj = b1[j] + rx * W1[3 * HIDW + j] + ry * W1[4 * HIDW + j]
                         + rz * W1[5 * HIDW + j];
        hj = swishf(hj);
        #pragma unroll
        for (int m = 0; m < MIDW; ++m) kacc[m] += hj * W2[j * MIDW + m];
    }
    #pragma unroll
    for (int m = 0; m < MIDW; ++m)
        lk[wave][l * MIDW + m] = (_Float16)swishf(kacc[m] + b2[m]);

    __syncthreads();

    // --- A operand: kern^T, ISA 16-bit A layout ---------------------------
    v16h a;
    {
        const int mrow  = lane & 15;
        const int kbase = (lane >= 16) ? 8 : 0;
        #pragma unroll
        for (int hh = 0; hh < 16; ++hh) {
            const int k = kbase + ((hh < 8) ? hh : hh + 8);
            a[hh] = lk[wave][k * MIDW + mrow];
        }
    }

    const int   nbr = idx[p * KNN + l];             // global feature row
    const float nm  = nmask[p * KNN + l];
    const float* srow = src + (size_t)nbr * src_stride;

    _Float16* drow = xpen + (size_t)(p - chunk_base) * (MIDW * cin);
    const int nmid = (lane >= 16) ? 8 : 0;
    const int ncol = lane & 15;

    const int ntiles = cin >> 4;
    for (int ct = 0; ct < ntiles; ++ct) {
        // --- B operand: masked neighbor features (lane = N col, elems = K table order) ---
        v16h bm;
        #pragma unroll
        for (int h = 0; h < 16; ++h) {
            const int kk = nmid + ((h < 8) ? h : h + 8);
            const int nbk = idx[p * KNN + kk];
            const float nmk = nmask[p * KNN + kk];
            float v = src[(size_t)nbk * src_stride + ct * 16 + ncol];
            v = relu_on ? fmaxf(v, 0.0f) : v;
            bm[h] = (_Float16)(v * nmk);
        }

        v8f c = {};
        c = __builtin_amdgcn_wmma_f32_16x16x32_f16(
                false, a, false, bm, (short)0, c, false, false);

        // --- D tile (C/D layout: vgpr r -> M=r / M=8+r) --------------------
        #pragma unroll
        for (int r = 0; r < 8; ++r)
            drow[(nmid + r) * cin + ct * 16 + ncol] = (_Float16)c[r];
    }
}

// ---------------------------------------------------------------------------
// Wl GEMM: out[pts, cout] = (xpen[pts, 16*cin] @ Wl + bl) / cnt, written into
// the DenseNet feature buffer at channel offset wr_off.
// All 8 waves of a block share one 16-col B panel: it is staged into LDS with
// double-buffered async global->LDS copies (256 thr x B64 = 2 KB/panel); the
// next panel's copy overlaps the current panel's WMMAs.
// ---------------------------------------------------------------------------
__global__ __launch_bounds__(256)
void liedn_wl_gemm_kernel(const _Float16* __restrict__ xpen, int Kdim,
                          const float* __restrict__ Wl,
                          const float* __restrict__ bl,
                          const float* __restrict__ cnt,
                          float* __restrict__ dst,
                          int cout, int wr_off, int chunk_base)
{
    __shared__ float bpanel[2][32 * 16];        // double-buffered f32 k-panel

    const int tid  = threadIdx.x;
    const int wave = tid >> 5;
    const int lane = tid & 31;
    const int rt   = blockIdx.x * 8 + wave;     // 16-row tile within chunk
    const int colt = blockIdx.y;                // 16-col tile

    // async-copy mapping: thread t moves 8B; krow = t/8, seg = t%8
    const int krow = tid >> 3;
    const int seg  = tid & 7;
    const float* gsrc0 = Wl + (size_t)krow * cout + colt * 16 + seg * 2;
    const unsigned lds0 =
        (unsigned)(uintptr_t)&bpanel[0][krow * 16 + seg * 2];
    const unsigned lds1 =
        (unsigned)(uintptr_t)&bpanel[1][krow * 16 + seg * 2];

    const int mrow = lane & 15;
    const int koff = (lane >= 16) ? 8 : 0;
    const _Float16* arow = xpen + (size_t)(rt * 16 + mrow) * Kdim + koff;

    const int nk = Kdim >> 5;                   // k-steps of 32

    async_copy_b64(lds0, gsrc0);                // prologue: panel 0 -> buf 0

    v8f acc = {};
    for (int ks = 0; ks < nk; ++ks) {
        const int k0 = ks << 5;
        if (ks + 1 < nk) {
            // prefetch next panel into the other buffer, then retire oldest
            async_copy_b64((ks & 1) ? lds0 : lds1,
                           gsrc0 + (size_t)(k0 + 32) * cout);
            asm volatile("s_wait_asynccnt 1" ::: "memory");
        } else {
            asm volatile("s_wait_asynccnt 0" ::: "memory");
        }
        __syncthreads();        // panel[ks&1] complete & visible block-wide

        // A: xpen f16, ISA A layout; halves 0..7 / 8..15 are contiguous runs
        const v8h alo = *(const v8h*)(arow + k0);
        const v8h ahi = *(const v8h*)(arow + k0 + 16);
        v16h a;
        #pragma unroll
        for (int h = 0; h < 8; ++h) { a[h] = alo[h]; a[h + 8] = ahi[h]; }

        // B col n = lane%16 from the LDS panel, k in table order (f32 -> f16)
        v16h bm;
        #pragma unroll
        for (int h = 0; h < 16; ++h)
            bm[h] = (_Float16)bpanel[ks & 1][(koff + ((h < 8) ? h : h + 8)) * 16 + (lane & 15)];

        acc = __builtin_amdgcn_wmma_f32_16x16x32_f16(
                  false, a, false, bm, (short)0, acc, false, false);

        __syncthreads();        // all waves done reading before re-fill
    }

    const int n   = lane & 15;
    const int mb  = (lane >= 16) ? 8 : 0;
    const float blv = bl[colt * 16 + n];
    #pragma unroll
    for (int r = 0; r < 8; ++r) {
        const int grow = chunk_base + rt * 16 + mb + r;
        dst[(size_t)grow * NFMAX + wr_off + colt * 16 + n] =
            (acc[r] + blv) / cnt[grow];
    }
}

// ---------------------------------------------------------------------------
// Global masked mean pool + 224x2 classifier.
// ---------------------------------------------------------------------------
__global__ __launch_bounds__(256)
void liedn_pool_kernel(const float* __restrict__ feat,
                       const unsigned char* __restrict__ mask,
                       const float* __restrict__ Wc,
                       const float* __restrict__ bc,
                       float* __restrict__ out)
{
    __shared__ float smf[NPER];
    __shared__ float pooled[NFMAX];
    __shared__ float cntsh;

    const int b = blockIdx.x, t = threadIdx.x;
    for (int i = t; i < NPER; i += 256) smf[i] = mask[b * NPER + i] ? 1.0f : 0.0f;
    __syncthreads();
    if (t == 0) {
        float c = 0.0f;
        for (int n = 0; n < NPER; ++n) c += smf[n];
        cntsh = fmaxf(c, 1.0f);
    }
    __syncthreads();
    if (t < NFMAX) {
        float s = 0.0f;
        for (int n = 0; n < NPER; ++n) {
            const float v = feat[((size_t)(b * NPER + n)) * NFMAX + t];
            s += fmaxf(v, 0.0f) * smf[n];
        }
        pooled[t] = s / cntsh;
    }
    __syncthreads();
    if (t < 2) {
        float s = bc[t];
        for (int ch = 0; ch < NFMAX; ++ch) s += pooled[ch] * Wc[ch * 2 + t];
        out[b * 2 + t] = s;
    }
}

// ---------------------------------------------------------------------------
// Host side.  d_in layout (tree-flatten, dict keys sorted inside params):
//   [0] coords [4,2048,3] f32      [1] vals [4,2048,16] f32
//   [2] mask   [4,2048]   bool(u8) [3] Wc [224,2] f32   [4] bc [2] f32
//   [5 + 6*li + {0..5}] = layer li {W1,W2,Wl,b1,b2,bl}
// ---------------------------------------------------------------------------
extern "C" void kernel_launch(void* const* d_in, const int* in_sizes, int n_in,
                              void* d_out, int out_size, void* d_ws, size_t ws_size,
                              hipStream_t stream)
{
    (void)in_sizes; (void)n_in; (void)out_size; (void)ws_size;

    const float*         coords = (const float*)d_in[0];
    const float*         vals   = (const float*)d_in[1];
    const unsigned char* mask   = (const unsigned char*)d_in[2];
    const float*         Wc     = (const float*)d_in[3];
    const float*         bc     = (const float*)d_in[4];
    auto LP = [&](int li, int w) -> const float* {
        return (const float*)d_in[5 + li * 6 + w];   // 0:W1 1:W2 2:Wl 3:b1 4:b2 5:bl
    };

    // workspace carve (256B aligned)
    char* ws = (char*)d_ws; size_t off = 0;
    auto carve = [&](size_t bytes) -> void* {
        void* p = ws + off; off = (off + bytes + 255) & ~(size_t)255; return p;
    };
    float*     feat0  = (float*)    carve((size_t)NPTS * NFMAX * 4);
    float*     feat1  = (float*)    carve((size_t)NPTS * NFMAX * 4);
    float*     relb   = (float*)    carve((size_t)NPTS * KNN * 3 * 4);
    int*       idxb   = (int*)      carve((size_t)NPTS * KNN * 4);
    float*     nmaskb = (float*)    carve((size_t)NPTS * KNN * 4);
    float*     cntb   = (float*)    carve((size_t)NPTS * 4);
    _Float16*  xpen   = (_Float16*) carve((size_t)CHUNK * (MIDW * NFMAX) * 2);

    liedn_knn_kernel<<<NPTS / 128, 128, 0, stream>>>(coords, mask, relb, idxb,
                                                     nmaskb, cntb);

    // layer schedule (DenseNet concat offsets / transition ping-pong)
    struct LDesc { int cin, cout, wroff, sstride, relu; const float* src; float* dst; };
    LDesc L[15];
    float* cur = feat0; float* oth = feat1;
    L[0] = {16, 32, 0, 16, 0, vals, cur};
    int nf = 32, li = 1;
    for (int bi = 0; bi < 3; ++bi) {
        for (int k = 0; k < 4; ++k) {
            L[li] = {nf, 16, nf, NFMAX, 1, cur, cur};
            nf += 16; ++li;
        }
        if (bi != 2) {
            L[li] = {nf, nf, 0, NFMAX, 1, cur, oth};
            float* tmp = cur; cur = oth; oth = tmp; ++li;
        }
    }

    for (int i = 0; i < 15; ++i) {
        const LDesc& d = L[i];
        for (int c = 0; c < NPTS / CHUNK; ++c) {
            const int cb = c * CHUNK;
            liedn_pen_kernel<<<CHUNK / 8, 256, 0, stream>>>(
                d.src, d.sstride, d.cin, d.relu, relb, idxb, nmaskb,
                LP(i, 0), LP(i, 3), LP(i, 1), LP(i, 4), xpen, cb);
            dim3 gg(CHUNK / 128, d.cout / 16);
            liedn_wl_gemm_kernel<<<gg, 256, 0, stream>>>(
                xpen, 16 * d.cin, LP(i, 2), LP(i, 5), cntb, d.dst,
                d.cout, d.wroff, cb);
        }
    }

    liedn_pool_kernel<<<BATCH, 256, 0, stream>>>(cur, mask, Wc, bc, (float*)d_out);
}
